// Head_1752346656850
// MI455X (gfx1250) — matrix operand order repaired and checked by reading, not compiled, not run on hardware
//
#include <hip/hip_runtime.h>
#include <hip/hip_bf16.h>

// ---------------------------------------------------------------------------
// Causal attention head (B=32, S=1024, D=64) for gfx1250 (MI455X).
// f16 WMMA (v_wmma_f32_16x16x32_f16) with fp32 accumulation, flash-softmax
// in the exp2 domain, double-buffered async-to-LDS K/V staging (ASYNCcnt).
// ---------------------------------------------------------------------------

typedef __attribute__((ext_vector_type(16))) _Float16     v16h;
typedef __attribute__((ext_vector_type(8)))  float        v8f;
typedef __attribute__((ext_vector_type(4)))  unsigned int v4u;
typedef __attribute__((ext_vector_type(4)))  int          v4i;

#define Bsz 32
#define Ssz 1024
#define Dh  64
#define NEG_INF (-1e30f)

#if __has_builtin(__builtin_amdgcn_global_load_async_to_lds_b128)
#define HAVE_ASYNC_LDS 1
// Builtin signature (from hipcc diagnostic): (int4 AS1*, int4 AS3*, Imm, Imm)
typedef __attribute__((address_space(1))) v4i as1_v4i;
typedef __attribute__((address_space(3))) v4i as3_v4i;
#else
#define HAVE_ASYNC_LDS 0
#endif

__device__ __forceinline__ v8f wmma_f16(v16h a, v16h b, v8f c) {
  // D = A(16x32 f16) * B(32x16 f16) + C(16x16 f32)
  return __builtin_amdgcn_wmma_f32_16x16x32_f16(
      /*neg_a=*/false, a, /*neg_b=*/false, b,
      /*c_mod=*/(short)0, c, /*reuse_a=*/false, /*reuse_b=*/false);
}

union FragU { v16h v; v4u q[2]; };

// A fragment (16x32 f16) from a row-major matrix m[ld], tile at (r0, c0).
// ISA layout: lanes 0-15 row M=lane, K = c0+{0..7,16..23};
//             lanes 16-31 row M=lane-16, K = c0+{8..15,24..31}.
__device__ __forceinline__ v16h load_A16x32(const _Float16* m, int ld,
                                            int r0, int c0, int lane) {
  const int hs  = lane >> 4;
  const int row = r0 + (lane & 15);
  const _Float16* p = m + (size_t)row * ld + c0 + hs * 8;
  FragU f;
  f.q[0] = *(const v4u*)(p);        // K = c0+hs*8    .. +7
  f.q[1] = *(const v4u*)(p + 16);   // K = c0+16+hs*8 .. +7
  return f.v;
}

// B fragment (32x16 f16) where memory m (row-major, stride ld) holds B
// TRANSPOSED: m[n][k] == B[k][n].  Tile at (n0, k0).
// ISA layout: lane L holds column N = L&15, K = k0 + (L>>4)*8 + {0..7,16..23}.
__device__ __forceinline__ v16h load_B32x16_T(const _Float16* m, int ld,
                                              int n0, int k0, int lane) {
  const int hs  = lane >> 4;
  const int row = n0 + (lane & 15);
  const _Float16* p = m + (size_t)row * ld + k0 + hs * 8;
  FragU f;
  f.q[0] = *(const v4u*)(p);
  f.q[1] = *(const v4u*)(p + 16);
  return f.v;
}

// 16-lane butterfly reductions (wave32: two independent groups of 16).
__device__ __forceinline__ float rmax16(float v) {
  v = fmaxf(v, __shfl_xor(v, 1, 16));
  v = fmaxf(v, __shfl_xor(v, 2, 16));
  v = fmaxf(v, __shfl_xor(v, 4, 16));
  v = fmaxf(v, __shfl_xor(v, 8, 16));
  return v;
}
__device__ __forceinline__ float rsum16(float v) {
  v += __shfl_xor(v, 1, 16);
  v += __shfl_xor(v, 2, 16);
  v += __shfl_xor(v, 4, 16);
  v += __shfl_xor(v, 8, 16);
  return v;
}

// ---------------------------------------------------------------------------
// Async K/V tile staging: one 32-key tile per block (K: 32x64 f16 = 4 KB
// contiguous; V^T: 64x32 f16 = 4 KB, 64 strided rows).  128 threads x 2
// b128 chunks each per matrix.
// ---------------------------------------------------------------------------
__device__ __forceinline__ void issue_kv_tile(const _Float16* __restrict__ Kb,
                                              const _Float16* __restrict__ Vb,
                                              int kb, _Float16* ldsK,
                                              _Float16* ldsV, int tid) {
  // K tile: 4096 contiguous bytes at Kb + kb*Dh; thread copies 32 B.
  const _Float16* gk = Kb + (size_t)kb * Dh + tid * 16;
  _Float16* lk = ldsK + tid * 16;
  // V tile: chunk c (16 B) -> Vt row c>>2, col (c&3)*8 within tile.
  // Thread owns chunks 2*tid, 2*tid+1 (always same row -> contiguous 32 B).
  const int c0 = tid * 2;
  const _Float16* gv = Vb + (size_t)(c0 >> 2) * Ssz + kb + (c0 & 3) * 8;
  _Float16* lv = ldsV + c0 * 8;
#if HAVE_ASYNC_LDS
  __builtin_amdgcn_global_load_async_to_lds_b128(
      (as1_v4i*)const_cast<_Float16*>(gk), (as3_v4i*)lk, 0, 0);
  __builtin_amdgcn_global_load_async_to_lds_b128(
      (as1_v4i*)const_cast<_Float16*>(gk), (as3_v4i*)lk, 16, 0);
  __builtin_amdgcn_global_load_async_to_lds_b128(
      (as1_v4i*)const_cast<_Float16*>(gv), (as3_v4i*)lv, 0, 0);
  __builtin_amdgcn_global_load_async_to_lds_b128(
      (as1_v4i*)const_cast<_Float16*>(gv), (as3_v4i*)lv, 16, 0);
#else
  ((v4u*)lk)[0] = ((const v4u*)gk)[0];
  ((v4u*)lk)[1] = ((const v4u*)gk)[1];
  ((v4u*)lv)[0] = ((const v4u*)gv)[0];
  ((v4u*)lv)[1] = ((const v4u*)gv)[1];
#endif
}

__device__ __forceinline__ void wait_async_tiles() {
#if HAVE_ASYNC_LDS
#if __has_builtin(__builtin_amdgcn_s_wait_asynccnt)
  __builtin_amdgcn_s_wait_asynccnt(0);
#else
  asm volatile("s_wait_asynccnt 0" ::: "memory");
#endif
#endif
}

// ---------------------------------------------------------------------------
// Kernel 0: fp32 -> f16 conversion of x and the three weight matrices.
// ---------------------------------------------------------------------------
__global__ void convert_kernel(const float* __restrict__ x,
                               const float* __restrict__ wq,
                               const float* __restrict__ wk,
                               const float* __restrict__ wv,
                               _Float16* __restrict__ xh,
                               _Float16* __restrict__ wqh,
                               _Float16* __restrict__ wkh,
                               _Float16* __restrict__ wvh, int nx) {
  int i = blockIdx.x * blockDim.x + threadIdx.x;
  if (i < nx) xh[i] = (_Float16)x[i];
  if (i < Dh * Dh) {
    wqh[i] = (_Float16)wq[i];
    wkh[i] = (_Float16)wk[i];
    wvh[i] = (_Float16)wv[i];
  }
}

// ---------------------------------------------------------------------------
// Kernel 1: QKV projection via WMMA.  q = x @ W^T  (torch Linear convention).
// W row-major [out][in] already stores the GEMM B-matrix transposed.
// Q, K stored row-major f16 [S][64]; V stored TRANSPOSED f16 [64][S].
// ---------------------------------------------------------------------------
__device__ __forceinline__ void project_store(v16h a0, v16h a1,
                                              const _Float16* __restrict__ w,
                                              _Float16* __restrict__ dst,
                                              int ld_dst, bool transpose,
                                              int r0, int lane) {
  const int hs = lane >> 4, nIdx = lane & 15;
#pragma unroll
  for (int c = 0; c < 4; ++c) {
    const int n0 = c * 16;
    v16h b0 = load_B32x16_T(w, Dh, n0, 0, lane);
    v16h b1 = load_B32x16_T(w, Dh, n0, 32, lane);
    v8f acc = {};
    acc = wmma_f16(a0, b0, acc);
    acc = wmma_f16(a1, b1, acc);
#pragma unroll
    for (int r = 0; r < 8; ++r) {
      const int row = r0 + r + 8 * hs;
      const int col = n0 + nIdx;
      if (!transpose)
        dst[(size_t)row * ld_dst + col] = (_Float16)acc[r];
      else
        dst[(size_t)col * ld_dst + row] = (_Float16)acc[r];
    }
  }
}

__global__ __launch_bounds__(128) void proj_kernel(
    const _Float16* __restrict__ xh,
    const _Float16* __restrict__ wqh, const _Float16* __restrict__ wkh,
    const _Float16* __restrict__ wvh,
    _Float16* __restrict__ Qh, _Float16* __restrict__ Kh,
    _Float16* __restrict__ Vt) {
  const int wave = threadIdx.x >> 5;
  const int lane = threadIdx.x & 31;
  const int b    = blockIdx.y;
  const int r0   = blockIdx.x * 64 + wave * 16;

  const _Float16* xb = xh + (size_t)b * Ssz * Dh;
  v16h a0 = load_A16x32(xb, Dh, r0, 0,  lane);
  v16h a1 = load_A16x32(xb, Dh, r0, 32, lane);

  _Float16* Qb = Qh + (size_t)b * Ssz * Dh;   // [S][64]
  _Float16* Kb = Kh + (size_t)b * Ssz * Dh;   // [S][64]
  _Float16* Vb = Vt + (size_t)b * Dh * Ssz;   // [64][S] (transposed)

  project_store(a0, a1, wqh, Qb, Dh,  false, r0, lane);
  project_store(a0, a1, wkh, Kb, Dh,  false, r0, lane);
  project_store(a0, a1, wvh, Vb, Ssz, true,  r0, lane);
}

// ---------------------------------------------------------------------------
// Kernel 2: flash attention.  1 block = 64 queries of one batch, 4 waves;
// each wave owns a 16-query tile.  Key loop in 32-key steps up to the causal
// frontier (uniform trip count -> EXEC all-ones for WMMA).  K/V tiles are
// staged once per block into double-buffered LDS via async-to-LDS loads.
// ---------------------------------------------------------------------------
__global__ __launch_bounds__(128) void attn_kernel(
    const _Float16* __restrict__ Qh, const _Float16* __restrict__ Kh,
    const _Float16* __restrict__ Vt, float* __restrict__ out) {
  __shared__ __align__(16) _Float16 ldsK[2][32 * 64];  // 2 x 4 KB
  __shared__ __align__(16) _Float16 ldsV[2][64 * 32];  // 2 x 4 KB
  __shared__ __align__(16) _Float16 plds[4][16 * 32];  // per-wave P staging

  const int tid  = threadIdx.x;
  const int wave = tid >> 5;
  const int lane = tid & 31;
  const int b    = blockIdx.y;
  const int tile = blockIdx.x;
  const int q0   = tile * 64 + wave * 16;
  const int hs   = lane >> 4;
  const int nIdx = lane & 15;

  const _Float16* Qb = Qh + (size_t)b * Ssz * Dh;
  const _Float16* Kb = Kh + (size_t)b * Ssz * Dh;
  const _Float16* Vb = Vt + (size_t)b * Dh * Ssz;

  // Q A-fragments (persistent), contraction dim D=64 split into 2x K=32.
  v16h qa0 = load_A16x32(Qb, Dh, q0, 0,  lane);
  v16h qa1 = load_A16x32(Qb, Dh, q0, 32, lane);

  v8f o0 = {}, o1 = {}, o2 = {}, o3 = {};
  float m[8], l[8];
#pragma unroll
  for (int r = 0; r < 8; ++r) { m[r] = NEG_INF; l[r] = 0.0f; }

  // exp2-domain softmax: fold D^-0.5 * log2(e) into one scale.
  const float scale2 = 0.125f * 1.4426950408889634f;
  const int niter = 2 * (tile + 1);    // causal: skip fully-masked key blocks

  issue_kv_tile(Kb, Vb, 0, ldsK[0], ldsV[0], tid);

  for (int it = 0; it < niter; ++it) {
    const int kb  = it * 32;
    const int cur = it & 1;

    wait_async_tiles();     // this wave's portion of buf[cur] has landed
    __syncthreads();        // everyone's portion visible; prev reads done

    if (it + 1 < niter)     // uniform; stream next tile into the other buffer
      issue_kv_tile(Kb, Vb, kb + 32, ldsK[cur ^ 1], ldsV[cur ^ 1], tid);

    const _Float16* kt = ldsK[cur];   // [32][64]
    const _Float16* vt = ldsV[cur];   // [64][32]

    // ---- S = Q * K^T for 32 keys: two 16x16 fp32 tiles, 4 WMMAs ----
    v16h b00 = load_B32x16_T(kt, Dh, 0,  0,  lane);
    v16h b01 = load_B32x16_T(kt, Dh, 0,  32, lane);
    v16h b10 = load_B32x16_T(kt, Dh, 16, 0,  lane);
    v16h b11 = load_B32x16_T(kt, Dh, 16, 32, lane);
    v8f s0 = {}, s1 = {};
    s0 = wmma_f16(qa0, b00, s0);
    s0 = wmma_f16(qa1, b01, s0);
    s1 = wmma_f16(qa0, b10, s1);
    s1 = wmma_f16(qa1, b11, s1);

    // ---- scale + causal mask (lane-local selects, no divergence) ----
#pragma unroll
    for (int r = 0; r < 8; ++r) {
      const int qrow = q0 + r + 8 * hs;
      s0[r] = ((kb + nIdx)      <= qrow) ? s0[r] * scale2 : NEG_INF;
      s1[r] = ((kb + 16 + nIdx) <= qrow) ? s1[r] * scale2 : NEG_INF;
    }

    // ---- online softmax (fp32, exp2 domain) ----
    float alpha[8];
#pragma unroll
    for (int r = 0; r < 8; ++r) {
      float mx   = rmax16(fmaxf(s0[r], s1[r]));
      float mnew = fmaxf(m[r], mx);
      alpha[r]   = exp2f(m[r] - mnew);
      float p0   = exp2f(s0[r] - mnew);
      float p1   = exp2f(s1[r] - mnew);
      l[r] = l[r] * alpha[r] + rsum16(p0 + p1);
      m[r] = mnew;
      s0[r] = p0;
      s1[r] = p1;
    }
#pragma unroll
    for (int r = 0; r < 8; ++r) {
      o0[r] *= alpha[r]; o1[r] *= alpha[r];
      o2[r] *= alpha[r]; o3[r] *= alpha[r];
    }

    // ---- C-layout -> A-fragment layout via per-wave LDS tile ----
    _Float16* pl = plds[wave];
#pragma unroll
    for (int r = 0; r < 8; ++r) {
      const int row = r + 8 * hs;
      pl[row * 32 + nIdx]      = (_Float16)s0[r];
      pl[row * 32 + 16 + nIdx] = (_Float16)s1[r];
    }
    // Same-wave DS ops are in order (DScnt); compiler inserts the waits.
    v16h pa = load_A16x32(pl, 32, 0, 0, lane);

    // ---- O += P * V : 4 WMMAs over the 64 head dims ----
    v16h vb0 = load_B32x16_T(vt, 32, 0,  0, lane);
    v16h vb1 = load_B32x16_T(vt, 32, 16, 0, lane);
    v16h vb2 = load_B32x16_T(vt, 32, 32, 0, lane);
    v16h vb3 = load_B32x16_T(vt, 32, 48, 0, lane);
    o0 = wmma_f16(pa, vb0, o0);
    o1 = wmma_f16(pa, vb1, o1);
    o2 = wmma_f16(pa, vb2, o2);
    o3 = wmma_f16(pa, vb3, o3);
  }

  // ---- epilogue: normalize by l, store fp32 ----
  float* ob = out + ((size_t)b * Ssz + q0) * Dh;
#pragma unroll
  for (int r = 0; r < 8; ++r) {
    const int row = r + 8 * hs;
    const float inv = 1.0f / l[r];
    ob[(size_t)row * Dh + nIdx]      = o0[r] * inv;
    ob[(size_t)row * Dh + 16 + nIdx] = o1[r] * inv;
    ob[(size_t)row * Dh + 32 + nIdx] = o2[r] * inv;
    ob[(size_t)row * Dh + 48 + nIdx] = o3[r] * inv;
  }
}

// ---------------------------------------------------------------------------
// Host launcher.
// ---------------------------------------------------------------------------
extern "C" void kernel_launch(void* const* d_in, const int* in_sizes, int n_in,
                              void* d_out, int out_size, void* d_ws,
                              size_t ws_size, hipStream_t stream) {
  const float* x  = (const float*)d_in[0];
  const float* Wq = (const float*)d_in[1];
  const float* Wk = (const float*)d_in[2];
  const float* Wv = (const float*)d_in[3];
  float* out = (float*)d_out;

  const int nx = in_sizes[0];  // B*S*D = 2,097,152

  // Workspace layout (bytes):
  //   [0,4M)    xh  f16 [B][S][64]
  //   [4M,8M)   Qh  f16 [B][S][64]
  //   [8M,12M)  Kh  f16 [B][S][64]
  //   [12M,16M) Vt  f16 [B][64][S]  (V transposed)
  //   [16M,..)  Wq/Wk/Wv f16 (3 * 8 KB)
  char* ws = (char*)d_ws;
  _Float16* xh  = (_Float16*)(ws);
  _Float16* Qh  = (_Float16*)(ws + ((size_t)4  << 20));
  _Float16* Kh  = (_Float16*)(ws + ((size_t)8  << 20));
  _Float16* Vt  = (_Float16*)(ws + ((size_t)12 << 20));
  _Float16* Wqh = (_Float16*)(ws + ((size_t)16 << 20));
  _Float16* Wkh = Wqh + Dh * Dh;
  _Float16* Wvh = Wkh + Dh * Dh;

  convert_kernel<<<(nx + 255) / 256, 256, 0, stream>>>(
      x, Wq, Wk, Wv, xh, Wqh, Wkh, Wvh, nx);

  dim3 grid(Ssz / 64, Bsz);  // 16 x 32 blocks
  proj_kernel<<<grid, 128, 0, stream>>>(xh, Wqh, Wkh, Wvh, Qh, Kh, Vt);
  attn_kernel<<<grid, 128, 0, stream>>>(Qh, Kh, Vt, out);
}
